// Model_2542620639958
// MI455X (gfx1250) — matrix operand order repaired and checked by reading, not compiled, not run on hardware
//
#include <hip/hip_runtime.h>
#include <hip/hip_bf16.h>

typedef __attribute__((ext_vector_type(16))) __bf16 v16bf;
typedef __attribute__((ext_vector_type(8)))  float  v8f;

#define VOCAB 32000
#define SEQ   2048
#define DM    512
#define BM    128
#define BN    128
#define KC    32
#define SQRT_DM 22.62741699796952f

// ---------------------------------------------------------------------------
// Kernel 1: recover token ids from the one-hot matrix [VOCAB, SEQ] row-major.
// Exactly one element per column is 1.0, so exactly one thread writes ids[t].
// ---------------------------------------------------------------------------
__global__ void extract_ids_kernel(const float* __restrict__ FR,
                                   int* __restrict__ ids) {
    long g = (long)blockIdx.x * blockDim.x + threadIdx.x;   // g = v*SEQ + t
    if (g < (long)VOCAB * SEQ) {
        if (FR[g] > 0.5f) {
            int t = (int)(g & (SEQ - 1));
            int v = (int)(g >> 11);                          // SEQ == 2^11
            ids[t] = v;
        }
    }
}

// ---------------------------------------------------------------------------
// Kernel 2: E[k, n] = sqrt(512) * V[k, ids[n]]   (coalesced output, 4 MB)
// ---------------------------------------------------------------------------
__global__ void build_E_kernel(const float* __restrict__ V,
                               const int* __restrict__ ids,
                               float* __restrict__ E) {
    int g = blockIdx.x * blockDim.x + threadIdx.x;          // g = k*SEQ + n
    if (g < DM * SEQ) {
        int n = g & (SEQ - 1);
        int k = g >> 11;
        E[g] = SQRT_DM * V[k * VOCAB + ids[n]];
    }
}

// ---------------------------------------------------------------------------
// split f32 -> hi/lo bf16 (truncating split; residual captured exactly enough
// for ~2^-16 relative accuracy after hi*hi + hi*lo + lo*hi)
// ---------------------------------------------------------------------------
__device__ __forceinline__ void split2(float x, unsigned short& hi,
                                       unsigned short& lo) {
    unsigned u = __builtin_bit_cast(unsigned, x);
    unsigned h = u & 0xFFFF0000u;
    hi = (unsigned short)(h >> 16);
    float r = x - __builtin_bit_cast(float, h);
    lo = (unsigned short)(__builtin_bit_cast(unsigned, r) >> 16);
}

// ---------------------------------------------------------------------------
// Kernel 3: out[m, n] = sum_k V[k, m] * E[k, n]
//   A = V^T (M=32000, K=512), B = E (K=512, N=2048), out [32000, 2048].
//   Block: 128x128 tile, 256 threads = 8 waves (wave32), KC=32 per step.
//   Wave tile: 32(M) x 64(N) -> 2x4 accumulators v8f.
//   LDS holds hi/lo bf16 fragments pre-permuted into WMMA register order so
//   consumers issue contiguous 32B ds loads per fragment.
// ---------------------------------------------------------------------------
__launch_bounds__(256)
__global__ void gemm_kernel(const float* __restrict__ V,
                            const float* __restrict__ E,
                            float* __restrict__ out) {
    __shared__ __align__(32) unsigned short sA[2][8 * 512]; // [hi/lo][8 sub * 32 lane * 16 elem]
    __shared__ __align__(32) unsigned short sB[2][8 * 512];

    const int tid  = threadIdx.x;
    const int lane = tid & 31;
    const int wave = tid >> 5;
    const int wm   = wave >> 1;            // 0..3 -> M offset wm*32
    const int wn   = wave & 1;             // 0..1 -> N offset wn*64
    const int m_blk = blockIdx.y * BM;
    const int n_blk = blockIdx.x * BN;

    v8f acc[2][4];
#pragma unroll
    for (int i = 0; i < 2; ++i)
#pragma unroll
        for (int j = 0; j < 4; ++j) acc[i][j] = {};

    for (int k0 = 0; k0 < DM; k0 += KC) {
        if (k0 + KC < DM)   // pull next A chunk toward L2/L0 (global_prefetch_b8)
            __builtin_prefetch(&V[(long)(k0 + KC + (tid >> 7)) * VOCAB + m_blk + (tid & 127)], 0, 0);

        // ---- cooperative fill: A tile (BM x KC), fragment-order hi/lo bf16 ----
#pragma unroll 4
        for (int f = 0; f < 16; ++f) {
            int idx = f * 256 + tid;
            int m = idx & (BM - 1);
            int k = idx >> 7;                          // 0..31
            float a = V[(long)(k0 + k) * VOCAB + m_blk + m];
            unsigned short h, l; split2(a, h, l);
            // ISA 7.12.2 16-bit A 16x32 layout: lane = (m&15) + 16*((k>>3)&1)
            // elem e = ((k>>4)<<3) | (((k>>1)&3)<<1) | (k&1)
            int sm = m >> 4, r = m & 15;
            int e    = ((k >> 4) << 3) | (((k >> 1) & 3) << 1) | (k & 1);
            int half = (k >> 3) & 1;
            int off  = sm * 512 + (r + (half << 4)) * 16 + e;
            sA[0][off] = h;
            sA[1][off] = l;
        }
        // ---- cooperative fill: B tile (KC x BN) ----
#pragma unroll 4
        for (int f = 0; f < 16; ++f) {
            int idx = f * 256 + tid;
            int n = idx & (BN - 1);
            int k = idx >> 7;
            float b = E[(k0 + k) * SEQ + n_blk + n];
            unsigned short h, l; split2(b, h, l);
            // B 32x16: lane = c + 16*((k>>3)&1), elem = ((k>>4)<<3) | (k&7)
            int sn = n >> 4, c = n & 15;
            int off = sn * 512 + (c + (((k >> 3) & 1) << 4)) * 16 + (((k >> 4) << 3) | (k & 7));
            sB[0][off] = h;
            sB[1][off] = l;
        }
        __syncthreads();

        const v16bf* pAh = reinterpret_cast<const v16bf*>(sA[0]);
        const v16bf* pAl = reinterpret_cast<const v16bf*>(sA[1]);
        const v16bf* pBh = reinterpret_cast<const v16bf*>(sB[0]);
        const v16bf* pBl = reinterpret_cast<const v16bf*>(sB[1]);

        v16bf Ah[2], Al[2], Bh[4], Bl[4];
#pragma unroll
        for (int i = 0; i < 2; ++i) {
            int sub = wm * 2 + i;
            Ah[i] = pAh[sub * 32 + lane];
            Al[i] = pAl[sub * 32 + lane];
        }
#pragma unroll
        for (int j = 0; j < 4; ++j) {
            int sub = wn * 4 + j;
            Bh[j] = pBh[sub * 32 + lane];
            Bl[j] = pBl[sub * 32 + lane];
        }

        // split-bf16 product: hi*hi + hi*lo + lo*hi  (3 wmma per 16x16 tile)
#pragma unroll
        for (int i = 0; i < 2; ++i)
#pragma unroll
            for (int j = 0; j < 4; ++j) {
                acc[i][j] = __builtin_amdgcn_wmma_f32_16x16x32_bf16(
                    false, Ah[i], false, Bh[j], (short)0, acc[i][j], false, false);
                acc[i][j] = __builtin_amdgcn_wmma_f32_16x16x32_bf16(
                    false, Ah[i], false, Bl[j], (short)0, acc[i][j], false, false);
                acc[i][j] = __builtin_amdgcn_wmma_f32_16x16x32_bf16(
                    false, Al[i], false, Bh[j], (short)0, acc[i][j], false, false);
            }
        __syncthreads();
    }

    // ---- store: C/D layout VGPR r -> M = r + 8*(lane>=16), N = lane&15 ----
    const int half = lane >> 4;
    const int c    = lane & 15;
#pragma unroll
    for (int i = 0; i < 2; ++i)
#pragma unroll
        for (int j = 0; j < 4; ++j) {
            int Mb = m_blk + wm * 32 + i * 16 + (half << 3);
            int Nc = n_blk + wn * 64 + j * 16 + c;
#pragma unroll
            for (int r = 0; r < 8; ++r)
                out[(long)(Mb + r) * SEQ + Nc] = acc[i][j][r];
        }
}

// ---------------------------------------------------------------------------
// Inputs (setup_inputs order): 0=EN_onehot (unused by the output!),
// 1=FR_onehot_shifted [VOCAB,SEQ] f32, 2=V [DM,VOCAB] f32.
// Output: [VOCAB, SEQ] f32.
// Workspace: ids (8 KB) + E (4 MB).
// ---------------------------------------------------------------------------
extern "C" void kernel_launch(void* const* d_in, const int* in_sizes, int n_in,
                              void* d_out, int out_size, void* d_ws, size_t ws_size,
                              hipStream_t stream) {
    const float* FR = (const float*)d_in[1];
    const float* V  = (const float*)d_in[2];
    float* out = (float*)d_out;

    int*   ids = (int*)d_ws;
    float* E   = (float*)((char*)d_ws + 8192);

    long tot = (long)VOCAB * SEQ;
    extract_ids_kernel<<<(unsigned)((tot + 255) / 256), 256, 0, stream>>>(FR, ids);
    build_E_kernel<<<(DM * SEQ + 255) / 256, 256, 0, stream>>>(V, ids, E);

    dim3 grid(SEQ / BN, VOCAB / BM);   // 16 x 250
    gemm_kernel<<<grid, 256, 0, stream>>>(V, E, out);
}
